// SegmentedParallelPoincareMovingWindow_66649302499515
// MI455X (gfx1250) — matrix operand order repaired and checked by reading, not compiled, not run on hardware
//
#include <hip/hip_runtime.h>
#include <hip/hip_bf16.h>
#include <math.h>

// ---------------------------------------------------------------------------
// SegmentedParallelPoincareMovingWindow on MI455X (gfx1250, wave32, WMMA)
//
// Shapes: B=1024, LOOKBACK=672, C=32, L=24, E=128, N=28  -> rows M = B*N = 28672
// Per stream: z = (1/32) * xn_flat[M,768] @ Wflat[768,128] + bias_mean[128]
// with k = l*32 + c (matches the contiguous memory order of x per (b,n) row).
// Memory-bound (~425 MB @ 23.3 TB/s); GEMM done with v_wmma_f32_16x16x32_f16.
// ---------------------------------------------------------------------------

typedef _Float16 v16h __attribute__((ext_vector_type(16)));
typedef _Float16 v8h  __attribute__((ext_vector_type(8)));
typedef float    v8f  __attribute__((ext_vector_type(8)));

#define NROWS  28672            // B * N
#define LSEG   24
#define NCH    32
#define NEMB   128
#define KDIM   768              // LSEG * NCH
#define KSTEPS 24               // KDIM / 32
#define ETILES 8                // NEMB / 16
#define MTILE  16
#define APAD   8                // LDS row padding (halfs) to spread banks
#define MAXRAD 0.99999f         // (1 - 1e-5) / sqrt(curv=1)
#define WPACK_ELEMS (4 * KDIM * NEMB)   // 393216 f16 = 768 KB

// ---------------------------------------------------------------------------
// Kernel 1a: repack 4 weight tensors W[c][l][e] (f32) into f16 WMMA-B fragment
// order: [q][kk][et][lane][i] where k = kk*32 + (i>>3)*16 + (lane>>4)*8 + (i&7), e = et*16+(lane&15)
// so the GEMM kernel loads one contiguous 32B v16h per lane per k-step.
// ---------------------------------------------------------------------------
__global__ __launch_bounds__(256)
void pack_weights(const float* __restrict__ Wt, const float* __restrict__ Wc,
                  const float* __restrict__ Wf, const float* __restrict__ Wr,
                  _Float16* __restrict__ wpack)
{
    int idx = blockIdx.x * 256 + threadIdx.x;            // < 393216
    int i    = idx & 15;
    int lane = (idx >> 4) & 31;
    int et   = (idx >> 9) & 7;
    int kkq  = idx >> 12;                                // q*24 + kk
    int kk   = kkq % 24;
    int q    = kkq / 24;
    int k = kk * 32 + (i >> 3) * 16 + (lane >> 4) * 8 + (i & 7);
    int e = et * 16 + (lane & 15);
    int c = k & 31;
    int l = k >> 5;
    const float* W = (q == 0) ? Wt : (q == 1) ? Wc : (q == 2) ? Wf : Wr;
    wpack[idx] = (_Float16)W[(c * LSEG + l) * NEMB + e];
    (void)kk;
}

// Kernel 1b: bias_mean[q][e] = (1/32) * sum_c b_q[c][e]
__global__ void pack_bias(const float* __restrict__ bt, const float* __restrict__ bc,
                          const float* __restrict__ bf, const float* __restrict__ br,
                          float* __restrict__ biasMean)
{
    int t = blockIdx.x * blockDim.x + threadIdx.x;       // 512 total
    if (t >= 4 * NEMB) return;
    int q = t >> 7, e = t & 127;
    const float* b = (q == 0) ? bt : (q == 1) ? bc : (q == 2) ? bf : br;
    float s = 0.f;
    #pragma unroll
    for (int c = 0; c < NCH; ++c) s += b[c * NEMB + e];
    biasMean[t] = s * (1.f / 32.f);
}

// ---------------------------------------------------------------------------
// Kernel 2: fused normalize + WMMA GEMM + expmap0/projx.
// grid = (NROWS/16, 4 streams), block = 256 (8 waves; wave w owns e-tile w).
// ---------------------------------------------------------------------------
__global__ __launch_bounds__(256)
void seg_wmma_hyp(const float* __restrict__ x0, const float* __restrict__ x1,
                  const float* __restrict__ x2, const float* __restrict__ x3,
                  const _Float16* __restrict__ wpack,
                  const float* __restrict__ biasMean,
                  const float* __restrict__ esc,
                  float* __restrict__ out)
{
    __shared__ _Float16 As[MTILE][KDIM + APAD];   // ~24.8 KB normalized f16 A panel
    __shared__ float    norm2[MTILE];

    const int tid = threadIdx.x;
    const int q   = blockIdx.y;
    const float* xin = (q == 0) ? x0 : (q == 1) ? x1 : (q == 2) ? x2 : x3;
    const size_t rowbase = (size_t)blockIdx.x * MTILE;

    if (tid < MTILE) norm2[tid] = 0.f;

    // ---- Stage 1: per-(row,channel) normalization over L=24 (ddof=1) -------
    // 512 tasks; each wave covers one row with consecutive c -> coalesced-ish
    #pragma unroll
    for (int t = 0; t < 2; ++t) {
        const int task = tid + t * 256;
        const int row  = task >> 5;
        const int c    = task & 31;
        const float* src = xin + (rowbase + (size_t)row) * KDIM + c;
        float v[LSEG];
        float s1 = 0.f, s2 = 0.f;
        #pragma unroll
        for (int l = 0; l < LSEG; ++l) {
            float f = src[l * NCH];
            v[l] = f; s1 += f; s2 += f * f;
        }
        float mean = s1 * (1.f / 24.f);
        float var  = fmaxf((s2 - 24.f * mean * mean) * (1.f / 23.f), 0.f);
        float inv  = 1.f / (sqrtf(var) + 1e-6f);
        #pragma unroll
        for (int l = 0; l < LSEG; ++l)
            As[row][l * NCH + c] = (_Float16)((v[l] - mean) * inv);
    }
    __syncthreads();

    // ---- Stage 2: 24 x v_wmma_f32_16x16x32_f16 ----------------------------
    const int lane = tid & 31;
    const int et   = tid >> 5;          // wave id == e-tile
    const int m    = lane & 15;         // A row / D column index
    const int h    = lane >> 4;         // lane half

    v8f acc = {0.f, 0.f, 0.f, 0.f, 0.f, 0.f, 0.f, 0.f};
    const v16h* wp = reinterpret_cast<const v16h*>(wpack)
                   + (size_t)q * KSTEPS * ETILES * 32 + et * 32 + lane;

    #pragma unroll
    for (int kk = 0; kk < KSTEPS; ++kk) {
        const int k0 = kk * 32;
        // ISA 16-bit A 16x32 layout: lane-half h holds K = k0+h*8.. and k0+16+h*8..
        v8h lo = *reinterpret_cast<const v8h*>(&As[m][k0 + h * 8]);
        v8h hi = *reinterpret_cast<const v8h*>(&As[m][k0 + 16 + h * 8]);
        v16h a;
        #pragma unroll
        for (int i = 0; i < 8; ++i) { a[i] = lo[i]; a[8 + i] = hi[i]; }
        v16h b = wp[(size_t)kk * ETILES * 32];   // one 32B load, pre-swizzled B frag
        acc = __builtin_amdgcn_wmma_f32_16x16x32_f16(false, a, false, b,
                                                     (short)0, acc, false, false);
    }

    // ---- Stage 3: epilogue  h = u * min(tanh(|u|), MAX_R)/|u|,  u = z*s ----
    float s;
    { float xs = esc[0]; s = (xs > 20.f) ? xs : log1pf(expf(xs)); }  // softplus
    const float bias = biasMean[q * NEMB + et * 16 + m];
    const int mbase = h * 8;

    float u[8];
    #pragma unroll
    for (int r = 0; r < 8; ++r) {
        u[r] = (acc[r] * (1.f / 32.f) + bias) * s;
        float p = u[r] * u[r];
        // reduce over the 16 lanes sharing this row (stays inside lane half)
        p += __shfl_xor(p, 1, 32);
        p += __shfl_xor(p, 2, 32);
        p += __shfl_xor(p, 4, 32);
        p += __shfl_xor(p, 8, 32);
        if (m == 0) atomicAdd(&norm2[mbase + r], p);   // ds_add_f32 across waves
    }
    __syncthreads();

    #pragma unroll
    for (int r = 0; r < 8; ++r) {
        float n2  = fmaxf(norm2[mbase + r], 1e-15f);
        float nn  = sqrtf(n2);
        float fac = fminf(tanhf(nn), MAXRAD) / nn;
        size_t grow = rowbase + (size_t)(mbase + r);
        out[((size_t)q * NROWS + grow) * NEMB + et * 16 + m] = u[r] * fac;
    }
}

// ---------------------------------------------------------------------------
// Kernel 3: Mobius combination. One wave per row; lane holds a float4 (E=32*4).
// ---------------------------------------------------------------------------
__device__ __forceinline__ float wsum32(float v)
{
    v += __shfl_xor(v, 16, 32);
    v += __shfl_xor(v, 8, 32);
    v += __shfl_xor(v, 4, 32);
    v += __shfl_xor(v, 2, 32);
    v += __shfl_xor(v, 1, 32);
    return v;
}

__device__ __forceinline__ float4 mobius_scalar_mul(float4 x, float r)
{
    float n2 = wsum32(x.x * x.x + x.y * x.y + x.z * x.z + x.w * x.w);
    float n  = sqrtf(fmaxf(n2, 1e-15f));
    float nc = fminf(n, 1.f - 1e-7f);
    float a  = 0.5f * logf((1.f + nc) / (1.f - nc));    // arctanh
    float f  = tanhf(r * a) / n;
    return make_float4(x.x * f, x.y * f, x.z * f, x.w * f);
}

__device__ __forceinline__ float4 mobius_add(float4 x, float4 y)
{
    float xy = wsum32(x.x * y.x + x.y * y.y + x.z * y.z + x.w * y.w);
    float x2 = wsum32(x.x * x.x + x.y * x.y + x.z * x.z + x.w * x.w);
    float y2 = wsum32(y.x * y.x + y.y * y.y + y.z * y.z + y.w * y.w);
    float A   = 1.f + 2.f * xy + y2;
    float Bc  = 1.f - x2;
    float den = 1.f + 2.f * xy + x2 * y2;
    float id  = 1.f / fmaxf(den, 1e-15f);
    return make_float4((A * x.x + Bc * y.x) * id, (A * x.y + Bc * y.y) * id,
                       (A * x.z + Bc * y.z) * id, (A * x.w + Bc * y.w) * id);
}

__global__ __launch_bounds__(256)
void mobius_combine(float* __restrict__ buf, const float* __restrict__ mw)
{
    const int lane = threadIdx.x & 31;
    const int wv   = threadIdx.x >> 5;
    const size_t row = (size_t)blockIdx.x * 8 + wv;      // < NROWS

    float w[4];
    {
        float a = mw[0], b = mw[1], c = mw[2], d = mw[3];
        float mx = fmaxf(fmaxf(a, b), fmaxf(c, d));
        w[0] = expf(a - mx); w[1] = expf(b - mx);
        w[2] = expf(c - mx); w[3] = expf(d - mx);
        float ssum = w[0] + w[1] + w[2] + w[3];
        float inv  = 1.f / ssum;
        w[0] *= inv; w[1] *= inv; w[2] *= inv; w[3] *= inv;
    }

    const size_t off = row * NEMB + (size_t)lane * 4;
    float4 ht = *reinterpret_cast<const float4*>(buf + 0ull * NROWS * NEMB + off);
    float4 hc = *reinterpret_cast<const float4*>(buf + 1ull * NROWS * NEMB + off);
    float4 hf = *reinterpret_cast<const float4*>(buf + 2ull * NROWS * NEMB + off);
    float4 hr = *reinterpret_cast<const float4*>(buf + 3ull * NROWS * NEMB + off);

    float4 comb = mobius_scalar_mul(ht, w[0]);
    comb = mobius_add(comb, mobius_scalar_mul(hc, w[1]));
    comb = mobius_add(comb, mobius_scalar_mul(hf, w[2]));
    comb = mobius_add(comb, mobius_scalar_mul(hr, w[3]));

    // projx
    float n2 = wsum32(comb.x * comb.x + comb.y * comb.y +
                      comb.z * comb.z + comb.w * comb.w);
    float n  = sqrtf(fmaxf(n2, 1e-15f));
    float f  = fminf(1.f, MAXRAD / n);
    comb = make_float4(comb.x * f, comb.y * f, comb.z * f, comb.w * f);

    *reinterpret_cast<float4*>(buf + 4ull * NROWS * NEMB + off) = comb;
}

// ---------------------------------------------------------------------------
extern "C" void kernel_launch(void* const* d_in, const int* in_sizes, int n_in,
                              void* d_out, int out_size, void* d_ws, size_t ws_size,
                              hipStream_t stream)
{
    (void)in_sizes; (void)n_in; (void)out_size; (void)ws_size;

    const float* trend = (const float*)d_in[0];
    const float* scrs  = (const float*)d_in[1];
    const float* sfin  = (const float*)d_in[2];
    const float* resid = (const float*)d_in[3];
    const float* W_t = (const float*)d_in[4];  const float* b_t = (const float*)d_in[5];
    const float* W_c = (const float*)d_in[6];  const float* b_c = (const float*)d_in[7];
    const float* W_f = (const float*)d_in[8];  const float* b_f = (const float*)d_in[9];
    const float* W_r = (const float*)d_in[10]; const float* b_r = (const float*)d_in[11];
    const float* esc = (const float*)d_in[12];
    const float* mw  = (const float*)d_in[13];

    _Float16* wpack   = (_Float16*)d_ws;                              // 768 KB
    float*    biasM   = (float*)((char*)d_ws + (size_t)WPACK_ELEMS * 2);  // +2 KB
    float*    out     = (float*)d_out;   // [h_t|h_c|h_f|h_r|comb], each 28672x128 f32

    pack_weights<<<WPACK_ELEMS / 256, 256, 0, stream>>>(W_t, W_c, W_f, W_r, wpack);
    pack_bias<<<2, 256, 0, stream>>>(b_t, b_c, b_f, b_r, biasM);

    dim3 grid(NROWS / MTILE, 4);                                      // 1792 x 4
    seg_wmma_hyp<<<grid, 256, 0, stream>>>(trend, scrs, sfin, resid,
                                           wpack, biasM, esc, out);

    mobius_combine<<<NROWS / 8, 256, 0, stream>>>(out, mw);
}
